// LocationAwareAttention_90245852823653
// MI455X (gfx1250) — compile-verified, matrix operand order checked by reading, not run on hardware
//
#include <hip/hip_runtime.h>
#include <math.h>

#define Bsz 32
#define Tsz 2000
#define DEsz 512
#define DDsz 1024
#define Asz 128
#define FNsz 32
#define FSsz 16
#define KW 33          // conv kernel width = 2*FS+1
#define TTILE 128      // rows of T per workgroup
#define KC 64          // K-chunk staged in LDS
#define PITCH 72       // LDS row pitch in bf16 elements (16B-aligned, conflict-spread)

typedef float v8f   __attribute__((ext_vector_type(8)));
typedef __bf16 bf8v  __attribute__((ext_vector_type(8)));
typedef __bf16 bf16v __attribute__((ext_vector_type(16)));
typedef unsigned short us8v __attribute__((ext_vector_type(8)));

__device__ __forceinline__ unsigned f2bf(float x) {
    union { float f; unsigned u; } c; c.f = x;
    unsigned r = c.u + 0x7FFFu + ((c.u >> 16) & 1u);   // round-to-nearest-even
    return r >> 16;
}
__device__ __forceinline__ unsigned pk2bf(float a, float b) {
    return f2bf(a) | (f2bf(b) << 16);
}

// one K=64 chunk of WMMA: A tile [TTILE x KC], B tile [Asz x KC] (row-major over n),
// all 8 B fragments prefetched into registers before the 8-WMMA chain.
__device__ __forceinline__ void mma_chunk(
    const unsigned short (&A)[TTILE][PITCH],
    const unsigned short (&Bm)[Asz][PITCH],
    v8f (&acc)[8], int rowA, int l16, int half)
{
    #pragma unroll
    for (int ks = 0; ks < KC / 32; ++ks) {
        // 16-bit A layout: lanes 0-15 row m, K 0-7 / 16-23; lanes 16-31 K 8-15 / 24-31
        const int base = ks * 32 + half * 8;
        us8v alo = *(const us8v*)&A[rowA][base];
        us8v ahi = *(const us8v*)&A[rowA][base + 16];
        us8v blo[8], bhi[8];
        #pragma unroll
        for (int nt = 0; nt < 8; ++nt) {
            int nr = nt * 16 + l16;
            blo[nt] = *(const us8v*)&Bm[nr][base];
            bhi[nt] = *(const us8v*)&Bm[nr][base + 16];
        }
        bf16v av = __builtin_shufflevector(
            __builtin_bit_cast(bf8v, alo), __builtin_bit_cast(bf8v, ahi),
            0,1,2,3,4,5,6,7,8,9,10,11,12,13,14,15);
        #pragma unroll
        for (int nt = 0; nt < 8; ++nt) {
            bf16v bv = __builtin_shufflevector(
                __builtin_bit_cast(bf8v, blo[nt]), __builtin_bit_cast(bf8v, bhi[nt]),
                0,1,2,3,4,5,6,7,8,9,10,11,12,13,14,15);
#if defined(__gfx1250__)
            acc[nt] = __builtin_amdgcn_wmma_f32_16x16x32_bf16(
                false, av, false, bv, (short)0, acc[nt], false, false);
#endif
        }
    }
}

// ---------------- tiny helper kernels ----------------

__global__ void zero_kernel(float* p, int n) {
    int i = blockIdx.x * 256 + threadIdx.x;
    if (i < n) p[i] = 0.f;
}

// W_eff[n][j] = sum_f att_w[n][f] * conv_w[f][0][j]
__global__ void weff_kernel(const float* __restrict__ att_w,
                            const float* __restrict__ conv_w,
                            float* __restrict__ weff) {
    int idx = blockIdx.x * 256 + threadIdx.x;
    if (idx < Asz * KW) {
        int n = idx / KW, j = idx % KW;
        float s = 0.f;
        for (int f = 0; f < FNsz; ++f) s += att_w[n * FNsz + f] * conv_w[f * KW + j];
        weff[idx] = s;
    }
}

// pdec[b][n] = att_b[n] + dec_w[n,:] . input_dec[b,:]
__global__ void pdec_kernel(const float* __restrict__ dec_w,
                            const float* __restrict__ input_dec,
                            const float* __restrict__ att_b,
                            float* __restrict__ pdec) {
    int b = blockIdx.x;
    int n = threadIdx.x;          // 128 threads
    float s = att_b[n];
    const float* w = dec_w + (size_t)n * DDsz;
    const float* x = input_dec + (size_t)b * DDsz;
    for (int k = 0; k < DDsz; ++k) s += w[k] * x[k];
    pdec[b * Asz + n] = s;
}

// ---------------- main fused score kernel (all-WMMA) ----------------
// grid: (ceil(T/TTILE), B), block: 256 (8 waves). Wave w owns M-rows [w*16, w*16+16)
// of this T-tile and all 8 N-tiles of A=128.
// Accumulators are seeded by the location-feature GEMM (prev_att windows x W_eff,
// K=33 zero-padded to 64), then extended by the enc x enc_w^T GEMM over K=512.

__global__ __launch_bounds__(256) void score_kernel(
    const float* __restrict__ enc, const int* __restrict__ lengths,
    const float* __restrict__ prev_att, const float* __restrict__ enc_w,
    const float* __restrict__ weff, const float* __restrict__ pdec,
    const float* __restrict__ out_w, float* __restrict__ score)
{
    __shared__ __align__(16) unsigned short sA[TTILE][PITCH];  // A tile, bf16
    __shared__ __align__(16) unsigned short sB[Asz][PITCH];    // B tile, bf16
    __shared__ float sPrev[TTILE + 2 * FSsz];
    __shared__ float sScore[TTILE];

    const int tid  = threadIdx.x;
    const int b    = blockIdx.y;
    const int t0   = blockIdx.x * TTILE;
    const int wv   = tid >> 5;
    const int lane = tid & 31;
    const int l16  = lane & 15;
    const int half = lane >> 4;
    const int rowA = wv * 16 + l16;

    // prev_att slab (zero-padded both ends), score accumulator
    for (int i = tid; i < TTILE + 2 * FSsz; i += 256) {
        int t = t0 + i - FSsz;
        sPrev[i] = (t >= 0 && t < Tsz) ? prev_att[(size_t)b * Tsz + t] : 0.f;
    }
    if (tid < TTILE) sScore[tid] = 0.f;
    __syncthreads();

    v8f acc[8] = {};

    // ---- chunk L: location-feature GEMM, K = 33 taps zero-padded to 64
    // sA[m][j] = prevslab[m+j] (j<33), sB[n][j] = W_eff[n][j] (j<33), else 0
    for (int q = tid; q < TTILE * (KC / 2); q += 256) {
        int row = q / (KC / 2);
        int cp  = q % (KC / 2);
        int j0 = cp * 2, j1 = j0 + 1;
        float p0 = (j0 < KW) ? sPrev[row + j0] : 0.f;
        float p1 = (j1 < KW) ? sPrev[row + j1] : 0.f;
        *(unsigned*)&sA[row][cp * 2] = pk2bf(p0, p1);
    }
    for (int q = tid; q < Asz * (KC / 2); q += 256) {
        int n  = q / (KC / 2);
        int cp = q % (KC / 2);
        int j0 = cp * 2, j1 = j0 + 1;
        float w0 = (j0 < KW) ? weff[n * KW + j0] : 0.f;
        float w1 = (j1 < KW) ? weff[n * KW + j1] : 0.f;
        *(unsigned*)&sB[n][cp * 2] = pk2bf(w0, w1);
    }
    __syncthreads();
    mma_chunk(sA, sB, acc, rowA, l16, half);
    __syncthreads();

    // ---- chunks 0..7: enc x enc_w^T GEMM over K = 512
    for (int kc = 0; kc < DEsz / KC; ++kc) {
        #pragma unroll
        for (int i = 0; i < (TTILE * KC / 4) / 256; ++i) {   // 8 float4 per thread
            int q = tid + i * 256;
            int row = q >> 4;           // KC/4 = 16 float4 per row
            int cq  = q & 15;
            int t = t0 + row;
            float4 v = make_float4(0.f, 0.f, 0.f, 0.f);
            if (t < Tsz)
                v = *(const float4*)(enc + ((size_t)b * Tsz + t) * DEsz + kc * KC + cq * 4);
            *(uint2*)&sA[row][cq * 4] = make_uint2(pk2bf(v.x, v.y), pk2bf(v.z, v.w));
        }
        #pragma unroll
        for (int i = 0; i < (Asz * KC / 4) / 256; ++i) {
            int q = tid + i * 256;
            int row = q >> 4;
            int cq  = q & 15;
            float4 v = *(const float4*)(enc_w + (size_t)row * DEsz + kc * KC + cq * 4);
            *(uint2*)&sB[row][cq * 4] = make_uint2(pk2bf(v.x, v.y), pk2bf(v.z, v.w));
        }
        __syncthreads();
        mma_chunk(sA, sB, acc, rowA, l16, half);
        __syncthreads();
    }

    // ---- epilogue: tanh(acc + pdec) . out_w  -> reduce over n via LDS atomics
    const float* pd = pdec + b * Asz;
    float outw_r[8], pdec_r[8];
    #pragma unroll
    for (int nt = 0; nt < 8; ++nt) {
        int n = nt * 16 + l16;
        outw_r[nt] = out_w[n];
        pdec_r[nt] = pd[n];
    }
    #pragma unroll
    for (int v = 0; v < 8; ++v) {
        int m = wv * 16 + half * 8 + v;    // C/D layout: lanes 0-15 M=v, 16-31 M=8+v
        float sv = 0.f;
        #pragma unroll
        for (int nt = 0; nt < 8; ++nt)
            sv += tanhf(acc[nt][v] + pdec_r[nt]) * outw_r[nt];
        atomicAdd(&sScore[m], sv);
    }
    __syncthreads();

    if (tid < TTILE) {
        int t = t0 + tid;
        if (t < Tsz) {
            float s = sScore[tid];
            if (t >= lengths[b]) s = -INFINITY;
            score[(size_t)b * Tsz + t] = s;
        }
    }
}

// ---------------- masked softmax over T per batch ----------------

__global__ __launch_bounds__(256) void softmax_kernel(const float* __restrict__ score,
                                                      float* __restrict__ att) {
    __shared__ float red[256];
    int b = blockIdx.x, tid = threadIdx.x;
    const float* s = score + (size_t)b * Tsz;
    float m = -INFINITY;
    for (int t = tid; t < Tsz; t += 256) m = fmaxf(m, s[t]);
    red[tid] = m; __syncthreads();
    for (int o = 128; o > 0; o >>= 1) {
        if (tid < o) red[tid] = fmaxf(red[tid], red[tid + o]);
        __syncthreads();
    }
    float mx = red[0]; __syncthreads();
    float e = 0.f;
    for (int t = tid; t < Tsz; t += 256) e += __expf(s[t] - mx);
    red[tid] = e; __syncthreads();
    for (int o = 128; o > 0; o >>= 1) {
        if (tid < o) red[tid] += red[tid + o];
        __syncthreads();
    }
    float inv = 1.f / red[0];
    for (int t = tid; t < Tsz; t += 256)
        att[(size_t)b * Tsz + t] = __expf(s[t] - mx) * inv;
}

// ---------------- context = sum_t att[b,t] * enc[b,t,:] ----------------
// grid (8 T-chunks, B); 256 threads each own 2 of 512 channels; atomicAdd into zeroed ctx.

__global__ __launch_bounds__(256) void context_kernel(const float* __restrict__ enc,
                                                      const float* __restrict__ att,
                                                      float* __restrict__ ctx) {
    int b = blockIdx.y;
    int c = blockIdx.x;
    int tid = threadIdx.x;
    int d = tid * 2;
    int tstart = c * (Tsz / 8), tend = tstart + (Tsz / 8);
    float a0 = 0.f, a1 = 0.f;
    for (int t = tstart; t < tend; ++t) {
        float w = att[(size_t)b * Tsz + t];
        const float2 e = *(const float2*)(enc + ((size_t)b * Tsz + t) * DEsz + d);
        a0 += w * e.x; a1 += w * e.y;
    }
    atomicAdd(&ctx[b * DEsz + d], a0);
    atomicAdd(&ctx[b * DEsz + d + 1], a1);
}

// ---------------- launcher ----------------

extern "C" void kernel_launch(void* const* d_in, const int* in_sizes, int n_in,
                              void* d_out, int out_size, void* d_ws, size_t ws_size,
                              hipStream_t stream) {
    const float* input_enc   = (const float*)d_in[0];
    const int*   enc_lengths = (const int*)  d_in[1];
    const float* input_dec   = (const float*)d_in[2];
    const float* prev_att    = (const float*)d_in[3];
    const float* conv_w      = (const float*)d_in[4];
    const float* enc_w       = (const float*)d_in[5];
    const float* dec_w       = (const float*)d_in[6];
    const float* att_w       = (const float*)d_in[7];
    const float* att_b       = (const float*)d_in[8];
    const float* out_w       = (const float*)d_in[9];

    float* ctx = (float*)d_out;                // [B, DE]
    float* att = ctx + Bsz * DEsz;             // [B, T]

    float* ws       = (float*)d_ws;
    float* ws_score = ws;                      // B*T
    float* ws_pdec  = ws + Bsz * Tsz;          // B*A
    float* ws_weff  = ws_pdec + Bsz * Asz;     // A*KW

    zero_kernel<<<(Bsz * DEsz + 255) / 256, 256, 0, stream>>>(ctx, Bsz * DEsz);
    weff_kernel<<<(Asz * KW + 255) / 256, 256, 0, stream>>>(att_w, conv_w, ws_weff);
    pdec_kernel<<<Bsz, Asz, 0, stream>>>(dec_w, input_dec, att_b, ws_pdec);

    dim3 g((Tsz + TTILE - 1) / TTILE, Bsz);
    score_kernel<<<g, 256, 0, stream>>>(input_enc, enc_lengths, prev_att, enc_w,
                                        ws_weff, ws_pdec, out_w, ws_score);

    softmax_kernel<<<Bsz, 256, 0, stream>>>(ws_score, att);
    context_kernel<<<dim3(8, Bsz), 256, 0, stream>>>(input_enc, att, ctx);
}
